// LRNet_CIFAR10_ver2_36009005810025
// MI455X (gfx1250) — matrix operand order repaired and checked by reading, not compiled, not run on hardware
//
#include <hip/hip_runtime.h>
#include <hip/hip_bf16.h>
#include <math.h>

// ---------------------------------------------------------------------------
// CDNA5 (gfx1250) LR-Net forward pass.
// - all heavy math on v_wmma_f32_16x16x32_bf16 (wave32 WMMA, f32 accum)
// - NHWC activations: K-slices contiguous -> interior conv tiles fetched by
//   the Tensor Data Mover (tensor_load_to_lds + s_wait_tensorcnt)
// ---------------------------------------------------------------------------

typedef __attribute__((ext_vector_type(16))) __bf16 v16bf;
typedef __attribute__((ext_vector_type(8)))  __bf16 v8bf;
typedef __attribute__((ext_vector_type(8)))  float  v8f;
typedef __attribute__((ext_vector_type(4)))  unsigned int uint32x4;
typedef __attribute__((ext_vector_type(8)))  int  int32x8;
typedef __attribute__((ext_vector_type(4)))  int  int32x4;

#define EPS_VAR 1e-10f
#define BN_EPS  1e-5f

__device__ __forceinline__ unsigned short f2bf(float f) {
  union { float f; unsigned u; } x; x.f = f;
  unsigned r = x.u + 0x7fffu + ((x.u >> 16) & 1u);   // round-to-nearest-even
  return (unsigned short)(r >> 16);
}
__device__ __forceinline__ float bf2f(unsigned short h) {
  union { unsigned u; float f; } x; x.u = ((unsigned)h) << 16;
  return x.f;
}

union Frag {
  v16bf v;
  v8bf  h[2];
  unsigned short s[16];
};

// A-matrix 16x32 bf16 fragment (ISA 7.12.2): m = lane%16;
// lanes 0-15 hold K 0..7 / 16..23, lanes 16-31 hold K 8..15 / 24..31.
// Caller passes p = row + kk + ((lane<16)?0:8).
__device__ __forceinline__ void loadA(Frag& f, const unsigned short* p) {
  f.h[0] = *(const v8bf*)(p);
  f.h[1] = *(const v8bf*)(p + 16);
}
// B-matrix 32x16 bf16 fragment: column n = lane%16; same K order as A:
// lanes 0-15 hold K 0..7 / 16..23, lanes 16-31 hold K 8..15 / 24..31.
__device__ __forceinline__ void loadB(Frag& f, const unsigned short* p) {
  f.h[0] = *(const v8bf*)(p);
  f.h[1] = *(const v8bf*)(p + 16);
}
__device__ __forceinline__ void squareFrag(Frag& d, const Frag& src) {
#pragma unroll
  for (int i = 0; i < 16; ++i) {
    float x = bf2f(src.s[i]);
    d.s[i] = f2bf(x * x);
  }
}
__device__ __forceinline__ void onesFrag(Frag& d) {
#pragma unroll
  for (int i = 0; i < 16; ++i) d.s[i] = 0x3F80u;   // bf16 1.0
}
__device__ __forceinline__ v8f wmma_bf16(const Frag& a, const Frag& b, v8f c) {
  return __builtin_amdgcn_wmma_f32_16x16x32_bf16(
      false, a.v, false, b.v, (short)0, c, false, false);
}

// ---------------------------------------------------------------------------
// Ternary weight statistics -> bf16 GEMM weights [Cout][Kpad],
// K = (kh*3+kw)*Cin + cin, zero padded to Kpad.
//   ew = E[w] ; wv = Var[w] (lrconv) or E[w]^2 (ver2) ; w2 = E[w^2]
// ---------------------------------------------------------------------------
__global__ __launch_bounds__(256) void wstats_kernel(
    const float* __restrict__ alpha, const float* __restrict__ betta,
    unsigned short* __restrict__ ew, unsigned short* __restrict__ wv,
    unsigned short* __restrict__ w2, int Cout, int Cin, int Kpad, int ver2)
{
  long long idx = (long long)blockIdx.x * blockDim.x + threadIdx.x;
  long long total = (long long)Cout * Kpad;
  if (idx >= total) return;
  int oc = (int)(idx / Kpad);
  int k  = (int)(idx - (long long)oc * Kpad);
  float e = 0.f, vv = 0.f, e2 = 0.f;
  if (k < 9 * Cin) {
    int tap = k / Cin;
    int ci  = k - tap * Cin;
    int kh = tap / 3, kw = tap - kh * 3;
    int src = ((oc * Cin + ci) * 3 + kh) * 3 + kw;
    float p0 = 1.f / (1.f + __expf(-alpha[src]));
    float p1 = (1.f - p0) / (1.f + __expf(-betta[src]));
    e  = 2.f * p1 - (1.f - p0);
    e2 = 1.f - p0;
    vv = ver2 ? (e * e) : (e2 - e * e);
  }
  ew[idx] = f2bf(e);
  wv[idx] = f2bf(vv);
  w2[idx] = f2bf(e2);
}

__global__ __launch_bounds__(256) void f32_to_bf16_kernel(
    const float* __restrict__ in, unsigned short* __restrict__ outp, long long n)
{
  long long i = (long long)blockIdx.x * blockDim.x + threadIdx.x;
  if (i < n) outp[i] = f2bf(in[i]);
}

// NCHW f32 -> NHWC bf16 (for the network input)
__global__ __launch_bounds__(256) void nchw_to_nhwc_kernel(
    const float* __restrict__ in, unsigned short* __restrict__ outp,
    int C, int HW, long long n)
{
  long long i = (long long)blockIdx.x * blockDim.x + threadIdx.x;
  if (i >= n) return;
  int chw = C * HW;
  int nimg = (int)(i / chw);
  int rem  = (int)(i - (long long)nimg * chw);
  int r = rem / C;
  int c = rem - r * C;
  outp[i] = f2bf(in[((size_t)nimg * C + c) * HW + r]);
}

// fc1 weights: absorb NCHW-flatten -> NHWC-flatten permutation of K=8192
// out[o][r*512 + c] = in[o][c*16 + r]
__global__ __launch_bounds__(256) void fc1w_permute_kernel(
    const float* __restrict__ in, unsigned short* __restrict__ outp)
{
  long long i = (long long)blockIdx.x * blockDim.x + threadIdx.x;
  if (i >= 1024LL * 8192LL) return;
  int o = (int)(i >> 13);
  int t = (int)(i & 8191);
  int r = t >> 9;           // 0..15
  int c = t & 511;          // 0..511
  outp[i] = f2bf(in[(size_t)o * 8192 + c * 16 + r]);
}

// e_x = erf(m * rsqrt(2*(v+eps)))
__global__ __launch_bounds__(256) void ex_kernel(
    const float* __restrict__ m, const float* __restrict__ v,
    unsigned short* __restrict__ ex, long long n)
{
  long long i = (long long)blockIdx.x * blockDim.x + threadIdx.x;
  if (i < n) ex[i] = f2bf(erff(m[i] * rsqrtf(2.f * (v[i] + EPS_VAR))));
}

// s = m + sqrt(v+eps)*noise ; m,v,s are NHWC, noise is NCHW
__global__ __launch_bounds__(256) void sample_kernel(
    const float* __restrict__ m, const float* __restrict__ v,
    const float* __restrict__ eps, float* __restrict__ s,
    int C, int HW, long long n)
{
  long long i = (long long)blockIdx.x * blockDim.x + threadIdx.x;
  if (i >= n) return;
  int chw = C * HW;
  int nimg = (int)(i / chw);
  int rem  = (int)(i - (long long)nimg * chw);
  int r = rem / C;
  int c = rem - r * C;
  s[i] = m[i] + sqrtf(v[i] + EPS_VAR) * eps[((size_t)nimg * C + c) * HW + r];
}

// per-channel mean / rsqrt(var) over NHWC tensor
__global__ __launch_bounds__(256) void bnstats_kernel(
    const float* __restrict__ S, float* __restrict__ mu, float* __restrict__ rs,
    int C, int NHW)
{
  __shared__ float s1[256], s2[256];
  int c = blockIdx.x;
  float a = 0.f, b = 0.f;
  for (int i = threadIdx.x; i < NHW; i += 256) {
    float x = S[(size_t)i * C + c];
    a += x; b += x * x;
  }
  s1[threadIdx.x] = a; s2[threadIdx.x] = b;
  __syncthreads();
  for (int st = 128; st > 0; st >>= 1) {
    if (threadIdx.x < st) {
      s1[threadIdx.x] += s1[threadIdx.x + st];
      s2[threadIdx.x] += s2[threadIdx.x + st];
    }
    __syncthreads();
  }
  if (threadIdx.x == 0) {
    float m = s1[0] / (float)NHW;
    float var = s2[0] / (float)NHW - m * m;
    mu[c] = m;
    rs[c] = rsqrtf(var + BN_EPS);
  }
}

__global__ __launch_bounds__(256) void bnapply_relu_kernel(
    const float* __restrict__ S, const float* __restrict__ mu,
    const float* __restrict__ rs, const float* __restrict__ g,
    const float* __restrict__ be, unsigned short* __restrict__ outp,
    int C, long long n)
{
  long long i = (long long)blockIdx.x * blockDim.x + threadIdx.x;
  if (i >= n) return;
  int c = (int)(i % C);
  float v = g[c] * (S[i] - mu[c]) * rs[c] + be[c];
  outp[i] = f2bf(fmaxf(v, 0.f));
}

// ---------------------------------------------------------------------------
// Fused implicit-GEMM LR-conv (NHWC). Block = 16 output pixels x 128 oc
// (4 waves x 2 M-tiles). Interior tiles: B tile (16px x 32 cin = regular
// 2D/3D tile in NHWC) DMA'd to LDS by the Tensor Data Mover with LDS padding
// reproducing the 80B bank-spread row stride. Border tiles / Cin=3: masked
// manual gather.
//   !VER2: M = ew*x         ; V = var_w*x^2
//    VER2: M = ew*e_x       ; V = e_w2*mask - ew^2*e_x^2   (mask==1 interior)
// ---------------------------------------------------------------------------
template<int STRIDE, bool VER2>
__global__ __launch_bounds__(128) void lrconv_wmma_kernel(
    const unsigned short* __restrict__ X,    // bf16 NHWC [B,H,W,Cin]
    const unsigned short* __restrict__ Wm,   // bf16 [Cout,Kpad] E[w]
    const unsigned short* __restrict__ Wv,   // bf16 Var[w] or E[w]^2
    const unsigned short* __restrict__ W2,   // bf16 E[w^2]  (VER2)
    const float* __restrict__ bias,
    float* __restrict__ Mout, float* __restrict__ Vout,  // f32 NHWC
    int Cin, int H, int W, int Cout, int OH, int OW, int Kpad)
{
  __shared__ __align__(64) unsigned short Bx[16][40];  // 80B row stride
  __shared__ unsigned short Bm[16][40];                // pad mask (border)
  __shared__ int pn[16], pihb[16], piwb[16], ppix[16], pmask[16];
  __shared__ int tapall;   // bit t: tap t in-bounds for all 16 pixels

  const int tid  = threadIdx.x;
  const int wave = tid >> 5;
  const int lane = tid & 31;
  const int ocbase = blockIdx.y * 128 + wave * 32;
  const int HW = OH * OW;

  if (tid < 16) {
    int p = blockIdx.x * 16 + tid;
    int n = p / HW;
    int r = p - n * HW;
    int oh = r / OW;
    int ow = r - oh * OW;
    pn[tid]   = n;
    pihb[tid] = oh * STRIDE - 1;   // pad = 1
    piwb[tid] = ow * STRIDE - 1;
    ppix[tid] = r;
    int msk = 0;
#pragma unroll
    for (int t = 0; t < 9; ++t) {
      int ih = oh * STRIDE - 1 + t / 3;
      int iw = ow * STRIDE - 1 + t % 3;
      if (ih >= 0 && ih < H && iw >= 0 && iw < W) msk |= (1 << t);
    }
    pmask[tid] = msk;
  }
  __syncthreads();
  if (tid == 0) {
    int a = 0x1FF;
    for (int i = 0; i < 16; ++i) a &= pmask[i];
    tapall = a;
  }
  __syncthreads();

  // uniform tile-origin scalars for the TDM descriptor
  const int p0tile = blockIdx.x * 16;
  const int n0  = p0tile / HW;
  const int r0  = p0tile - n0 * HW;
  const int oh0 = r0 / OW;
  const int ow0 = r0 - oh0 * OW;
  const int R    = (OW < 16) ? OW : 16;   // pixels per output row in tile
  const int rows = 16 / R;

  v8f z = {0.f,0.f,0.f,0.f,0.f,0.f,0.f,0.f};
  v8f accM0 = z, accM1 = z, accV0 = z, accV1 = z, accO0 = z, accO1 = z;

  const int Ktot = 9 * Cin;
  const bool fast = ((Cin & 31) == 0);
  const int mrow  = lane & 15;
  const int abase = (lane < 16) ? 0 : 8;
  const int bcol  = lane & 15;
  const int bk0   = (lane < 16) ? 0 : 8;
  const unsigned short* wm0 = Wm + (size_t)(ocbase + mrow) * Kpad;
  const unsigned short* wm1 = wm0 + (size_t)16 * Kpad;
  const unsigned short* wv0 = Wv + (size_t)(ocbase + mrow) * Kpad;
  const unsigned short* wv1 = wv0 + (size_t)16 * Kpad;
  const unsigned short* w20 = W2 + (size_t)(ocbase + mrow) * Kpad;
  const unsigned short* w21 = w20 + (size_t)16 * Kpad;

  for (int kk = 0; kk < Ktot; kk += 32) {
    int tap0 = 0, cin0 = 0;
    if (fast) { tap0 = kk / Cin; cin0 = kk - tap0 * Cin; }
    const bool useTDM = fast && (((tapall >> tap0) & 1) != 0);

    if (useTDM) {
      // ---- Tensor Data Mover: one DMA for the whole 16x32 B tile ----
      if (wave == 0) {
        const int kh0 = tap0 / 3, kw0 = tap0 - kh0 * 3;
        const long long ih0 = (long long)oh0 * STRIDE - 1 + kh0;
        const long long iw0 = (long long)ow0 * STRIDE - 1 + kw0;
        const unsigned long long gaddr = (unsigned long long)(uintptr_t)X +
            2ull * ((((unsigned long long)n0 * H + ih0) * W + iw0) * Cin + cin0);
        const unsigned ldsoff = (unsigned)(uintptr_t)(&Bx[0][0]);
        const long long st0 = (long long)STRIDE * Cin;       // pixel stride
        const long long st1 = (long long)STRIDE * W * Cin;   // out-row stride

        uint32x4 g0;
        g0[0] = 1u;                                   // count=1, normal mode
        g0[1] = ldsoff;                               // LDS byte address
        g0[2] = (unsigned)(gaddr & 0xFFFFFFFFull);
        g0[3] = (unsigned)((gaddr >> 32) & 0x1FFFFFFull) | (2u << 30); // type=2

        int32x8 g1;
        g1[0] = (1 << 16)            // data_size = 2 bytes
              | (1 << 20)            // pad_enable
              | (3 << 22)            // pad_interval: every 16 DWORDs (64B)
              | (3 << 25);           // pad_amount: 4 DWORDs (16B) -> 80B rows
        g1[1] = (int)0xFFFF0000u;                     // tensor_dim0 lo16 (max)
        g1[2] = (int)0xFFFF7FFFu;                     // dim0 hi | dim1 lo (max)
        g1[3] = (int)(0x7FFFu | (32u << 16));         // dim1 hi | tile_dim0=32
        g1[4] = R | ((rows > 1 ? rows : 0) << 16);    // tile_dim1 | tile_dim2
        g1[5] = (int)(st0 & 0xFFFFFFFFll);            // dim0_stride lo32
        g1[6] = (int)(((st0 >> 32) & 0xFFFFll) | ((st1 & 0xFFFFll) << 16));
        g1[7] = (int)((st1 >> 16) & 0xFFFFFFFFll);    // dim1_stride hi32

        int32x4 g2 = {0x7FFFFFFF, 0, 0, 0};           // tensor_dim2 (max)
        int32x4 g3 = {0, 0, 0, 0};
        int32x8 g4 = {0, 0, 0, 0, 0, 0, 0, 0};        // unused trailing group
        __builtin_amdgcn_tensor_load_to_lds(g0, g1, g2, g3, g4, 0);
        __builtin_amdgcn_s_wait_tensorcnt(0);
      }
    } else {
      // ---- masked manual gather (borders, and the Cin=3 first layer) ----
      for (int j = tid; j < 512; j += 128) {
        int p  = j >> 5;
        int ki = j & 31;
        int kg = kk + ki;
        float val = 0.f;
        unsigned short msk = 0;
        if (kg < Ktot) {
          int tap, ci;
          if (fast) { tap = tap0; ci = cin0 + ki; }
          else      { tap = kg / Cin; ci = kg - tap * Cin; }
          int kh = tap / 3, kw = tap - kh * 3;
          int ih = pihb[p] + kh;
          int iw = piwb[p] + kw;
          if (ih >= 0 && ih < H && iw >= 0 && iw < W) {
            msk = 0x3F80u;
            val = bf2f(X[(((size_t)pn[p] * H + ih) * W + iw) * Cin + ci]);
          }
        }
        Bx[p][ki] = f2bf(val);
        if (VER2) Bm[p][ki] = msk;
      }
    }
    __syncthreads();

    Frag b;  loadB(b, &Bx[bcol][bk0]);
    Frag b2; squareFrag(b2, b);
    Frag a;
    loadA(a, wm0 + kk + abase); accM0 = wmma_bf16(a, b,  accM0);
    loadA(a, wm1 + kk + abase); accM1 = wmma_bf16(a, b,  accM1);
    loadA(a, wv0 + kk + abase); accV0 = wmma_bf16(a, b2, accV0);
    loadA(a, wv1 + kk + abase); accV1 = wmma_bf16(a, b2, accV1);
    if (VER2) {
      Frag bm;
      if (useTDM) onesFrag(bm);           // interior: mask == 1 everywhere
      else        loadB(bm, &Bm[bcol][bk0]);
      loadA(a, w20 + kk + abase); accO0 = wmma_bf16(a, bm, accO0);
      loadA(a, w21 + kk + abase); accO1 = wmma_bf16(a, bm, accO1);
    }
    __syncthreads();
  }

  // ---- epilogue: C/D layout N=lane%16, M = i + (lane<16?0:8); NHWC out ----
  const int ncol = lane & 15;
  const int n = pn[ncol];
  const int r = ppix[ncol];
  const int mofs = (lane < 16) ? 0 : 8;
  const size_t rowbase = ((size_t)n * HW + r) * Cout;
#pragma unroll
  for (int t = 0; t < 2; ++t) {
    const v8f& aM = t ? accM1 : accM0;
    const v8f& aV = t ? accV1 : accV0;
    const v8f& aO = t ? accO1 : accO0;
    int ocb = ocbase + t * 16 + mofs;
    float mo[8], vo[8];
#pragma unroll
    for (int i = 0; i < 8; ++i) {
      mo[i] = aM[i] + bias[ocb + i];
      vo[i] = VER2 ? (aO[i] - aV[i]) : aV[i];
    }
    *(float4*)(Mout + rowbase + ocb)     = make_float4(mo[0], mo[1], mo[2], mo[3]);
    *(float4*)(Mout + rowbase + ocb + 4) = make_float4(mo[4], mo[5], mo[6], mo[7]);
    *(float4*)(Vout + rowbase + ocb)     = make_float4(vo[0], vo[1], vo[2], vo[3]);
    *(float4*)(Vout + rowbase + ocb + 4) = make_float4(vo[4], vo[5], vo[6], vo[7]);
  }
}

// ---------------------------------------------------------------------------
// Dense GEMM: Out[n,m] = Act[n,:] . Wt[m,:] + bias[m]; K-contiguous bf16.
// Pure global-load fragments. Optional relu / bf16 output.
// ---------------------------------------------------------------------------
template<bool RELU, bool OBF16>
__global__ __launch_bounds__(128) void gemm_wmma_kernel(
    const unsigned short* __restrict__ Wt,   // [M,K]
    const unsigned short* __restrict__ Act,  // [N,K]
    const float* __restrict__ bias,
    void* __restrict__ OutP,                 // [N,M]
    int M, int N, int K)
{
  const int lane = threadIdx.x & 31;
  const int wave = threadIdx.x >> 5;
  const int ocbase = blockIdx.y * 64 + wave * 16;
  const int nbase  = blockIdx.x * 16;

  int mrow = ocbase + (lane & 15);
  if (mrow > M - 1) mrow = M - 1;              // clamp (stores are guarded)
  const int abase = (lane < 16) ? 0 : 8;
  const int ncol  = lane & 15;
  const int bk0   = (lane < 16) ? 0 : 8;
  const unsigned short* wrow = Wt  + (size_t)mrow * K;
  const unsigned short* arow = Act + (size_t)(nbase + ncol) * K;

  v8f acc = {0.f,0.f,0.f,0.f,0.f,0.f,0.f,0.f};
  for (int kk = 0; kk < K; kk += 32) {
    Frag a, b;
    loadA(a, wrow + kk + abase);
    loadB(b, arow + kk + bk0);
    acc = wmma_bf16(a, b, acc);
  }

  const int nn = nbase + ncol;
  const int mofs = (lane < 16) ? 0 : 8;
  (void)N;
#pragma unroll
  for (int i = 0; i < 8; ++i) {
    int oc = ocbase + mofs + i;
    if (oc < M) {
      float v = acc[i] + bias[oc];
      if (RELU) v = fmaxf(v, 0.f);
      if (OBF16) ((unsigned short*)OutP)[(size_t)nn * M + oc] = f2bf(v);
      else       ((float*)OutP)[(size_t)nn * M + oc] = v;
    }
  }
}

// ---------------------------------------------------------------------------
static inline unsigned gblk(long long n) { return (unsigned)((n + 255) / 256); }

extern "C" void kernel_launch(void* const* d_in, const int* in_sizes, int n_in,
                              void* d_out, int out_size, void* d_ws, size_t ws_size,
                              hipStream_t stream)
{
  (void)in_sizes; (void)n_in; (void)out_size; (void)ws_size;

  const float* x = (const float*)d_in[0];
  const float *Al[7], *Bl[7], *Cl[7];
  for (int l = 1; l <= 6; ++l) {
    Al[l] = (const float*)d_in[1 + 3 * (l - 1)];
    Bl[l] = (const float*)d_in[2 + 3 * (l - 1)];
    Cl[l] = (const float*)d_in[3 + 3 * (l - 1)];
  }
  const float* g3   = (const float*)d_in[19];
  const float* be3  = (const float*)d_in[20];
  const float* g6   = (const float*)d_in[21];
  const float* be6  = (const float*)d_in[22];
  const float* fc1w = (const float*)d_in[23];
  const float* fc1b = (const float*)d_in[24];
  const float* fc2w = (const float*)d_in[25];
  const float* fc2b = (const float*)d_in[26];
  const float* eps3 = (const float*)d_in[27];
  const float* eps6 = (const float*)d_in[28];
  float* out = (float*)d_out;

  const int Bn = 256;
  const int Cin[7]  = {0,   3, 128, 128, 256, 256, 512};
  const int Co[7]   = {0, 128, 128, 256, 256, 512, 512};
  const int Kp[7]   = {0,  32, 1152, 1152, 2304, 2304, 4608};
  const int v2f[7]  = {0,   0,   1,   1,   0,   1,   1};

  // ---- workspace arena (~550 MB, static offsets) ----
  size_t off = 0;
  auto alloc = [&](size_t bytes) -> char* {
    char* p = (char*)d_ws + off;
    off = (off + bytes + 255) & ~(size_t)255;
    return p;
  };
  unsigned short *ew[7], *wv[7], *w2[7];
  for (int l = 1; l <= 6; ++l) {
    size_t nb = (size_t)Co[l] * Kp[l] * 2;
    ew[l] = (unsigned short*)alloc(nb);
    wv[l] = (unsigned short*)alloc(nb);
    w2[l] = (unsigned short*)alloc(nb);
  }
  unsigned short* xb = (unsigned short*)alloc((size_t)Bn * 3 * 32 * 32 * 2);
  float* m1 = (float*)alloc((size_t)33554432 * 4);
  float* v1 = (float*)alloc((size_t)33554432 * 4);
  float* m2 = (float*)alloc((size_t)8388608 * 4);
  float* v2 = (float*)alloc((size_t)8388608 * 4);
  unsigned short* act1 = (unsigned short*)alloc((size_t)33554432 * 2);
  unsigned short* act2 = (unsigned short*)alloc((size_t)16777216 * 2);
  float* sbuf = (float*)alloc((size_t)16777216 * 4);
  unsigned short* fc1wb = (unsigned short*)alloc((size_t)1024 * 8192 * 2);
  unsigned short* fc2wb = (unsigned short*)alloc((size_t)10 * 1024 * 2);
  float* mu = (float*)alloc(512 * 4);
  float* rs = (float*)alloc(512 * 4);

  for (int l = 1; l <= 6; ++l) {
    long long tot = (long long)Co[l] * Kp[l];
    wstats_kernel<<<gblk(tot), 256, 0, stream>>>(
        Al[l], Bl[l], ew[l], wv[l], w2[l], Co[l], Cin[l], Kp[l], v2f[l]);
  }
  nchw_to_nhwc_kernel<<<gblk(786432), 256, 0, stream>>>(x, xb, 3, 1024, 786432LL);
  fc1w_permute_kernel<<<gblk(8388608), 256, 0, stream>>>(fc1w, fc1wb);
  f32_to_bf16_kernel<<<gblk(10240), 256, 0, stream>>>(fc2w, fc2wb, 10240LL);

  // L1: lrconv 3->128, s1, 32x32  (Cin=3: manual-gather path)
  lrconv_wmma_kernel<1, false><<<dim3(16384, 1), 128, 0, stream>>>(
      xb, ew[1], wv[1], w2[1], Cl[1], m1, v1, 3, 32, 32, 128, 32, 32, Kp[1]);
  long long n1 = (long long)Bn * 128 * 1024;
  ex_kernel<<<gblk(n1), 256, 0, stream>>>(m1, v1, act1, n1);

  // L2: ver2 128->128, s2 -> 16x16
  lrconv_wmma_kernel<2, true><<<dim3(4096, 1), 128, 0, stream>>>(
      act1, ew[2], wv[2], w2[2], Cl[2], m2, v2, 128, 32, 32, 128, 16, 16, Kp[2]);
  long long n2 = (long long)Bn * 128 * 256;
  ex_kernel<<<gblk(n2), 256, 0, stream>>>(m2, v2, act2, n2);

  // L3: ver2 128->256, s1, 16x16
  lrconv_wmma_kernel<1, true><<<dim3(4096, 2), 128, 0, stream>>>(
      act2, ew[3], wv[3], w2[3], Cl[3], m1, v1, 128, 16, 16, 256, 16, 16, Kp[3]);
  long long n3 = (long long)Bn * 256 * 256;
  sample_kernel<<<gblk(n3), 256, 0, stream>>>(m1, v1, eps3, sbuf, 256, 256, n3);
  bnstats_kernel<<<256, 256, 0, stream>>>(sbuf, mu, rs, 256, Bn * 256);
  bnapply_relu_kernel<<<gblk(n3), 256, 0, stream>>>(sbuf, mu, rs, g3, be3, act1, 256, n3);

  // L4: lrconv 256->256, s2 -> 8x8
  lrconv_wmma_kernel<2, false><<<dim3(1024, 2), 128, 0, stream>>>(
      act1, ew[4], wv[4], w2[4], Cl[4], m2, v2, 256, 16, 16, 256, 8, 8, Kp[4]);
  long long n4 = (long long)Bn * 256 * 64;
  ex_kernel<<<gblk(n4), 256, 0, stream>>>(m2, v2, act2, n4);

  // L5: ver2 256->512, s1, 8x8
  lrconv_wmma_kernel<1, true><<<dim3(1024, 4), 128, 0, stream>>>(
      act2, ew[5], wv[5], w2[5], Cl[5], m1, v1, 256, 8, 8, 512, 8, 8, Kp[5]);
  long long n5 = (long long)Bn * 512 * 64;
  ex_kernel<<<gblk(n5), 256, 0, stream>>>(m1, v1, act1, n5);

  // L6: ver2 512->512, s2 -> 4x4
  lrconv_wmma_kernel<2, true><<<dim3(256, 4), 128, 0, stream>>>(
      act1, ew[6], wv[6], w2[6], Cl[6], m2, v2, 512, 8, 8, 512, 4, 4, Kp[6]);
  long long n6 = (long long)Bn * 512 * 16;
  sample_kernel<<<gblk(n6), 256, 0, stream>>>(m2, v2, eps6, sbuf, 512, 16, n6);
  bnstats_kernel<<<512, 256, 0, stream>>>(sbuf, mu, rs, 512, Bn * 16);
  bnapply_relu_kernel<<<gblk(n6), 256, 0, stream>>>(sbuf, mu, rs, g6, be6, act2, 512, n6);

  // FC1: [256,8192] x [1024,8192]^T + b, relu -> bf16 act1
  gemm_wmma_kernel<true, true><<<dim3(16, 16), 128, 0, stream>>>(
      fc1wb, act2, fc1b, act1, 1024, 256, 8192);
  // FC2: [256,1024] x [10,1024]^T + b -> f32 logits
  gemm_wmma_kernel<false, false><<<dim3(16, 1), 128, 0, stream>>>(
      fc2wb, act1, fc2b, out, 10, 256, 1024);
}
